// CrossCorrAngleEstimator_32530082300136
// MI455X (gfx1250) — matrix operand order repaired and checked by reading, not compiled, not run on hardware
//
#include <hip/hip_runtime.h>
#include <math.h>

// CDNA5 / gfx1250: wave32, WMMA f32 16x16x32 f16.
typedef __attribute__((ext_vector_type(16))) _Float16 v16h;
typedef __attribute__((ext_vector_type(8)))  float    v8f;

#define WIN   4800
#define NLAG  41
#define MAXD  20
#define S0    320   // WIN/16 = 300 phase-columns, padded to 10 chunks of 32
#define R1    328   // a1 phase-row stride (2 lead pad + 300 + tail pad)
#define NKK   10    // k-chunks of 32 along the 320 s-dimension

// One wave per batch item.
// Polyphase decomposition: t = 16 s + p.
//   B[k,n]   = a0[16 s + n]              (s = 32 kk + k)
//   A_j[m,k] = a1[16 (s - (j-2)) + m]    (tile j -> lag base 16 (j-2))
//   C_j[m,n] = sum over t = n (mod 16) of a0[t] * a1[t - (16(j-2)+n-m)]
// Each (lag in [-20,20], phase) pair lives in exactly one tile cell.
__global__ __launch_bounds__(32) void xcorr_wmma(const float* __restrict__ audio,
                                                 float* __restrict__ out,
                                                 int batch) {
  __shared__ _Float16 h0t[16 * S0];   // a0, phase-major: [p][s]
  __shared__ _Float16 h1t[16 * R1];   // a1, phase-major with guards: [p][2 + s]
  __shared__ float    cscr[40 * 32];  // accumulator spill: [(j*8+v)][lane]

  const int b    = blockIdx.x;
  const int lane = threadIdx.x;
  const float* a0 = audio + (size_t)b * (2 * WIN);
  const float* a1 = a0 + WIN;

  // ---- Phase 1: zero guards, then stream-convert f32 -> f16 into LDS ----
  for (int i = lane; i < 16 * 20; i += 32) {           // h0t tail pad s in [300,320)
    int p = i / 20, s = 300 + i % 20;
    h0t[p * S0 + s] = (_Float16)0.0f;
  }
  for (int i = lane; i < 16 * 28; i += 32) {           // h1t guards s2 in [0,2)+[302,328)
    int p = i / 28, c = i % 28;
    int s2 = (c < 2) ? c : (300 + c);
    h1t[p * R1 + s2] = (_Float16)0.0f;
  }
  for (int t = lane; t < WIN; t += 32) {               // coalesced 128B/wave global reads
    int p = t & 15, s = t >> 4;
    h0t[p * S0 + s]     = (_Float16)a0[t];
    h1t[p * R1 + 2 + s] = (_Float16)a1[t];
  }
  __syncthreads();

  // ---- Phase 2: 10 k-chunks x 5 lag tiles of v_wmma_f32_16x16x32_f16 ----
  v8f acc[5] = {};
  const int mn = lane & 15;     // m (A row) == n (B col) for this lane
  const int hi = lane >> 4;     // lane half selects K sub-range
  for (int kk = 0; kk < NKK; ++kk) {
    // B fragment (lane half lo K=0-7,16-23; hi K=8-15,24-31; n = lane%16)
    v16h bf;
    const _Float16* bs = &h0t[mn * S0 + 32 * kk + 8 * hi];  // two 16B runs, 16B aligned
#pragma unroll
    for (int e = 0; e < 8; ++e) { bf[e] = bs[e]; bf[e + 8] = bs[16 + e]; }
#pragma unroll
    for (int j = 0; j < 5; ++j) {
      // A fragment (ISA: lane half lo K=0-7,16-23; hi K=8-15,24-31; m = lane%16)
      v16h af;
      const int base = mn * R1 + 2 + 32 * kk + 8 * hi - (j - 2);
#pragma unroll
      for (int e = 0; e < 8; ++e) {
        af[e]     = h1t[base + e];
        af[e + 8] = h1t[base + 16 + e];
      }
      acc[j] = __builtin_amdgcn_wmma_f32_16x16x32_f16(
          /*neg_a=*/false, af, /*neg_b=*/false, bf,
          /*c_mod=*/(short)0, acc[j], /*reuse_a=*/false, /*reuse_b=*/false);
    }
  }
  __syncthreads();

  // ---- Phase 3: spill C tiles, assemble 41 lags, top-2 |corr| ----
#pragma unroll
  for (int j = 0; j < 5; ++j)
#pragma unroll
    for (int v = 0; v < 8; ++v)
      cscr[(j * 8 + v) * 32 + lane] = acc[j][v];   // value (j, m=v+8*hi, n=lane&15)
  __syncthreads();

  float best1 = -1.0f, best2 = -1.0f;
  int bestL = 0;
  for (int L = 0; L < NLAG; ++L) {
    const int Lv = L - MAXD;
    float s = 0.0f;
#pragma unroll
    for (int p = 0; p < 16; ++p) {
      int q = p - Lv;                 // m = q mod 16, tile j = 2 + (m - q)/16
      int m = q & 15;
      int j = 2 + ((m - q) >> 4);
      s += cscr[(j * 8 + (m & 7)) * 32 + p + ((m >> 3) << 4)];  // uniform addr: LDS broadcast
    }
    float a = fabsf(s);
    if (a > best1) { best2 = best1; best1 = a; bestL = L; }   // first-occurrence argmax
    else if (a > best2) best2 = a;
  }

  if (lane == 0) {
    float lag = (float)(bestL - MAXD);
    float st  = lag * (343.0f / (0.1f * 48000.0f));
    st = fminf(1.0f, fmaxf(-1.0f, st));
    float theta = 90.0f - asinf(st) * (180.0f / 3.14159265358979323846f);
    float conf  = tanhf(0.5f * best1 / (best2 + 1e-9f));
    out[b]             = lag;    // optimal_lag (stored as float)
    out[batch + b]     = theta;  // theta
    out[2 * batch + b] = conf;   // confidence
  }
}

extern "C" void kernel_launch(void* const* d_in, const int* in_sizes, int n_in,
                              void* d_out, int out_size, void* d_ws, size_t ws_size,
                              hipStream_t stream) {
  const float* audio = (const float*)d_in[0];
  float* out = (float*)d_out;
  const int batch = in_sizes[0] / (2 * WIN);
  xcorr_wmma<<<dim3(batch), dim3(32), 0, stream>>>(audio, out, batch);
}
